// ClusterRefineNet_34574486733447
// MI455X (gfx1250) — compile-verified, matrix operand order checked by reading, not run on hardware
//
#include <hip/hip_runtime.h>
#include <hip/hip_bf16.h>

#define Bn 4
#define Nn 16384
#define Mn 1024
#define Cn 32
#define S0n 16
#define S1n 32
#define MATCH_R 0.6f

typedef __attribute__((ext_vector_type(16))) __bf16 v16bf;
typedef __attribute__((ext_vector_type(8)))  __bf16 v8bf;
typedef __attribute__((ext_vector_type(8)))  float  v8f;
typedef __attribute__((ext_vector_type(4)))  float  v4f;

__device__ __forceinline__ v8f wmma_bf16(v16bf a, v16bf b, v8f c) {
  return __builtin_amdgcn_wmma_f32_16x16x32_bf16(false, a, false, b, (short)0, c, false, false);
}

// A fragment (16x32 bf16) from row-major bf16 [rows][64] in LDS.
// lane l: row = mt*16 + (l&15); hi = l>>4; elems 0..7 -> K = ks*32+8*hi+e,
// elems 8..15 -> K = ks*32+8*hi+16+e  (CDNA5 ISA 16-bit A layout).
// Both halves are 16B-aligned contiguous -> 2x ds_load_b128.
__device__ __forceinline__ v16bf load_a_lds(const __bf16* h, int mt, int ks, int lane) {
  int row = mt * 16 + (lane & 15);
  int hi  = (lane >> 4) & 1;
  const v8bf* r = (const v8bf*)(h + row * 64 + ks * 32 + hi * 8);
  v16bf a;
  *(v8bf*)&a       = r[0];   // K octet 0
  *((v8bf*)&a + 1) = r[2];   // K octet +16
  return a;
}

// B fragment from pre-packed fragment-order weights: [frag][lane][16 bf16]
// -> one contiguous 32B load per lane (2x global_load_b128, coalesced).
__device__ __forceinline__ v16bf load_b_packed(const __bf16* __restrict__ P, int f, int lane) {
  return *(const v16bf*)(P + ((size_t)f * 32 + lane) * 16);
}

// ---------------- Kernel 1: pack weights into WMMA fragment order ----------------
__device__ __forceinline__ void pack_frag(const float* __restrict__ W, int Ksrc, int Nfull,
                                          int Ksteps, __bf16* __restrict__ out, int t) {
  int e    = t & 15;
  int lane = (t >> 4) & 31;
  int f    = t >> 9;                 // 512 elements per fragment
  int nt   = f / Ksteps;
  int ks   = f - nt * Ksteps;
  int n    = nt * 16 + (lane & 15);
  int hi   = lane >> 4;
  int k    = ks * 32 + 8 * hi + e + ((e >= 8) ? 8 : 0);
  out[t] = (__bf16)((k < Ksrc) ? W[k * Nfull + n] : 0.0f);
}

__global__ void __launch_bounds__(256) prep_weights_kernel(
    const float* __restrict__ w00, const float* __restrict__ w01,
    const float* __restrict__ w10, const float* __restrict__ w11,
    const float* __restrict__ fcw,
    __bf16* o00, __bf16* o01, __bf16* o10, __bf16* o11, __bf16* ofc) {
  int t = blockIdx.x * blockDim.x + threadIdx.x;
  if (t < 64 * 64)   { pack_frag(w00, 35, 64, 2, o00, t);   // 8 frags (4nt x 2ks)
                       pack_frag(w10, 35, 64, 2, o10, t); }
  if (t < 64 * 128)  { pack_frag(w01, 64, 128, 2, o01, t);  // 16 frags (8nt x 2ks)
                       pack_frag(w11, 64, 128, 2, o11, t); }
  if (t < 256 * 128) pack_frag(fcw, 256, 128, 8, ofc, t);   // 64 frags (8nt x 8ks)
}

// ---------------- Kernel 2: ordered ball query (wave32 scan) ----------------
__global__ void __launch_bounds__(128) ballquery_kernel(
    const float* __restrict__ xyz, const float* __restrict__ new_xyz,
    int* __restrict__ idx0, int* __restrict__ idx1) {
  __shared__ int lists[4][S0n + S1n];
  int lane = threadIdx.x & 31;
  int wave = __builtin_amdgcn_readfirstlane(threadIdx.x >> 5);
  int q = blockIdx.x * 4 + wave;          // [0, B*M)
  int b = q / Mn;
  const float* nx = new_xyz + (size_t)q * 3;
  float cx = nx[0], cy = nx[1], cz = nx[2];
  const float* pb = xyz + (size_t)b * Nn * 3;
  int* l0 = lists[wave];
  int* l1 = lists[wave] + S0n;
  const float r0sq = 0.15f * 0.15f, r1sq = 0.3f * 0.3f;
  int cnt0 = 0, cnt1 = 0;
  unsigned lt = (1u << lane) - 1u;
  for (int base = 0; base < Nn; base += 32) {
    int i = base + lane;
    float dx = pb[i * 3 + 0] - cx;
    float dy = pb[i * 3 + 1] - cy;
    float dz = pb[i * 3 + 2] - cz;
    float d2 = dx * dx + dy * dy + dz * dz;
    unsigned m1 = __builtin_amdgcn_ballot_w32(d2 < r1sq);
    unsigned m0 = __builtin_amdgcn_ballot_w32(d2 < r0sq);
    if (cnt1 < S1n && ((m1 >> lane) & 1u)) {
      int p = cnt1 + __popc(m1 & lt);
      if (p < S1n) l1[p] = i;
    }
    if (cnt0 < S0n && ((m0 >> lane) & 1u)) {
      int p = cnt0 + __popc(m0 & lt);
      if (p < S0n) l0[p] = i;
    }
    cnt1 = min(S1n, cnt1 + __popc(m1));
    cnt0 = min(S0n, cnt0 + __popc(m0));
    if (cnt0 >= S0n && cnt1 >= S1n) break;   // early exit (wave-uniform)
  }
  asm volatile("s_wait_dscnt 0" ::: "memory");
  int f0 = (cnt0 > 0) ? l0[0] : 0;
  int f1 = (cnt1 > 0) ? l1[0] : 0;
  if (lane < S0n) idx0[q * S0n + lane] = (lane < cnt0) ? l0[lane] : f0;
  idx1[q * S1n + lane] = (lane < cnt1) ? l1[lane] : f1;
}

// ---------------- Kernel 3: per-(query,scale) gather + MLP + max ----------------
__global__ void __launch_bounds__(128) sa_mlp_kernel(
    const float* __restrict__ xyz, const float* __restrict__ new_xyz,
    const float* __restrict__ features, const float* __restrict__ score,
    const int* __restrict__ idx0, const int* __restrict__ idx1,
    const __bf16* __restrict__ wp00, const __bf16* __restrict__ w01,
    const __bf16* __restrict__ wp10, const __bf16* __restrict__ w11,
    const float* __restrict__ g00, const float* __restrict__ b00,
    const float* __restrict__ g01, const float* __restrict__ b01,
    const float* __restrict__ g10, const float* __restrict__ b10,
    const float* __restrict__ g11, const float* __restrict__ b11,
    float* __restrict__ fbuf) {
  __shared__ __bf16 hbuf[4][32 * 64];       // per-wave S x 64 tile (bf16)
  int lane = threadIdx.x & 31;
  int wave = __builtin_amdgcn_readfirstlane(threadIdx.x >> 5);
  int unit = blockIdx.x * 4 + wave;         // [0, B*M*2), wave-uniform (SGPR)
  int scale = unit & 1;
  int q = unit >> 1;
  int b = q / Mn;
  int nmt = scale ? 2 : 1;                  // M-tiles (S = 16*nmt), SGPR-uniform
  int S = nmt << 4;
  const int*    nidx = scale ? (idx1 + q * S1n) : (idx0 + q * S0n);
  const __bf16* W0 = scale ? wp10 : wp00;   // packed frags, 64x64 (pad from 35x64)
  const __bf16* W1 = scale ? w11 : w01;     // packed frags, 64x128
  const float*  gA = scale ? g10 : g00;  const float* bA = scale ? b10 : b00;
  const float*  gB = scale ? g11 : g01;  const float* bB = scale ? b11 : b01;
  __bf16* h = hbuf[wave];

  // ---- gather: lane s builds row s = [rel_xyz(3) | fea*score(32) | 0-pad] ----
  float cx = new_xyz[(size_t)q * 3 + 0];
  float cy = new_xyz[(size_t)q * 3 + 1];
  float cz = new_xyz[(size_t)q * 3 + 2];
  if (lane < S) {
    int pi = nidx[lane];
    const float* p = xyz + ((size_t)b * Nn + pi) * 3;
    float sc = score[(size_t)b * Nn + pi];
    __bf16* row = h + lane * 64;
    row[0] = (__bf16)(p[0] - cx);
    row[1] = (__bf16)(p[1] - cy);
    row[2] = (__bf16)(p[2] - cz);
    const float* fp = features + ((size_t)b * Nn + pi) * Cn;
#pragma unroll
    for (int c = 0; c < Cn; ++c) row[3 + c] = (__bf16)(fp[c] * sc);
#pragma unroll
    for (int c = 3 + Cn; c < 64; ++c) row[c] = (__bf16)0.0f;
  }
  asm volatile("s_wait_dscnt 0" ::: "memory");

  int ncol = lane & 15, hi = lane >> 4;

  // ---- layer 0: (S x 64pad) @ (64 x 64), cache A frags, overwrite h with relu out ----
  v16bf a00 = load_a_lds(h, 0, 0, lane);
  v16bf a01 = load_a_lds(h, 0, 1, lane);
  v16bf a10 = {}, a11 = {};
  if (nmt == 2) { a10 = load_a_lds(h, 1, 0, lane); a11 = load_a_lds(h, 1, 1, lane); }
#pragma unroll
  for (int nt = 0; nt < 4; ++nt) {
    v16bf bf0 = load_b_packed(W0, nt * 2 + 0, lane);
    v16bf bf1 = load_b_packed(W0, nt * 2 + 1, lane);
    int n = nt * 16 + ncol;
    float gg = gA[n], bb = bA[n];
    v8f acc = {};
    acc = wmma_bf16(a00, bf0, acc);
    acc = wmma_bf16(a01, bf1, acc);
#pragma unroll
    for (int v = 0; v < 8; ++v)
      h[(v + 8 * hi) * 64 + n] = (__bf16)fmaxf(acc[v] * gg + bb, 0.0f);
    if (nmt == 2) {
      v8f acc2 = {};
      acc2 = wmma_bf16(a10, bf0, acc2);
      acc2 = wmma_bf16(a11, bf1, acc2);
#pragma unroll
      for (int v = 0; v < 8; ++v)
        h[(16 + v + 8 * hi) * 64 + n] = (__bf16)fmaxf(acc2[v] * gg + bb, 0.0f);
    }
  }
  asm volatile("s_wait_dscnt 0" ::: "memory");

  // ---- layer 1: (S x 64) @ (64 x 128), relu, max over samples ----
  v16bf c00 = load_a_lds(h, 0, 0, lane);
  v16bf c01 = load_a_lds(h, 0, 1, lane);
  v16bf c10 = {}, c11 = {};
  if (nmt == 2) { c10 = load_a_lds(h, 1, 0, lane); c11 = load_a_lds(h, 1, 1, lane); }
  float* fo = fbuf + (size_t)q * 256 + scale * 128;
#pragma unroll
  for (int nt = 0; nt < 8; ++nt) {
    v16bf bf0 = load_b_packed(W1, nt * 2 + 0, lane);
    v16bf bf1 = load_b_packed(W1, nt * 2 + 1, lane);
    int n = nt * 16 + ncol;
    float gg = gB[n], bb = bB[n];
    v8f acc = {};
    acc = wmma_bf16(c00, bf0, acc);
    acc = wmma_bf16(c01, bf1, acc);
    float mx = 0.0f;                               // relu floor
#pragma unroll
    for (int v = 0; v < 8; ++v) mx = fmaxf(mx, acc[v] * gg + bb);
    if (nmt == 2) {
      v8f acc2 = {};
      acc2 = wmma_bf16(c10, bf0, acc2);
      acc2 = wmma_bf16(c11, bf1, acc2);
#pragma unroll
      for (int v = 0; v < 8; ++v) mx = fmaxf(mx, acc2[v] * gg + bb);
    }
    mx = fmaxf(mx, __shfl_xor(mx, 16, 32));        // combine row halves
    if (lane < 16) fo[nt * 16 + ncol] = mx;
  }
}

// ---------------- Kernel 4: FC 256->128 (WMMA) + 128->3 offset head ----------------
__global__ void __launch_bounds__(128) fc_kernel(
    const float* __restrict__ fbuf, const __bf16* __restrict__ fcw,
    const float* __restrict__ fcg, const float* __restrict__ fcb,
    const float* __restrict__ offw, const float* __restrict__ offb,
    const float* __restrict__ new_xyz, float* __restrict__ out) {
  __shared__ float xb[4][16 * 128];
  int lane = threadIdx.x & 31;
  int wave = __builtin_amdgcn_readfirstlane(threadIdx.x >> 5);
  int row0 = (blockIdx.x * 4 + wave) * 16;     // 16 query rows per wave
  int ncol = lane & 15, hi = lane >> 4;
  const float* arow = fbuf + (size_t)(row0 + ncol) * 256;
  v16bf af[8];
#pragma unroll
  for (int ks = 0; ks < 8; ++ks) {
    const v4f* r = (const v4f*)(arow + ks * 32 + hi * 8);   // 32B aligned
    v4f x0 = r[0], x1 = r[1], x2 = r[4], x3 = r[5];
#pragma unroll
    for (int e = 0; e < 4; ++e) {
      af[ks][e]      = (__bf16)x0[e];
      af[ks][e + 4]  = (__bf16)x1[e];
      af[ks][e + 8]  = (__bf16)x2[e];
      af[ks][e + 12] = (__bf16)x3[e];
    }
  }
  float* xw = xb[wave];
#pragma unroll
  for (int nt = 0; nt < 8; ++nt) {
    v8f acc = {};
#pragma unroll
    for (int ks = 0; ks < 8; ++ks) {
      v16bf bf = load_b_packed(fcw, nt * 8 + ks, lane);
      acc = wmma_bf16(af[ks], bf, acc);
    }
    int n = nt * 16 + ncol;
    float gg = fcg[n], bb = fcb[n];
#pragma unroll
    for (int v = 0; v < 8; ++v)
      xw[(v + 8 * hi) * 128 + n] = fmaxf(acc[v] * gg + bb, 0.0f);
  }
  asm volatile("s_wait_dscnt 0" ::: "memory");
  for (int t = lane; t < 48; t += 32) {        // 16 rows x 3 outputs
    int r = t / 3, j = t % 3;
    const float* xr = xw + r * 128;
    float s = 0.0f;
#pragma unroll 4
    for (int c = 0; c < 128; ++c) s += xr[c] * offw[c * 3 + j];
    int q = row0 + r;
    out[(size_t)q * 3 + j] = (s + offb[j]) * MATCH_R + new_xyz[(size_t)q * 3 + j];
  }
}

extern "C" void kernel_launch(void* const* d_in, const int* in_sizes, int n_in,
                              void* d_out, int out_size, void* d_ws, size_t ws_size,
                              hipStream_t stream) {
  const float* xyz      = (const float*)d_in[0];
  const float* new_xyz  = (const float*)d_in[1];
  const float* features = (const float*)d_in[2];
  const float* score    = (const float*)d_in[3];
  const float* w0_0 = (const float*)d_in[4];
  const float* g0_0 = (const float*)d_in[5];
  const float* b0_0 = (const float*)d_in[6];
  const float* w0_1 = (const float*)d_in[7];
  const float* g0_1 = (const float*)d_in[8];
  const float* b0_1 = (const float*)d_in[9];
  const float* w1_0 = (const float*)d_in[10];
  const float* g1_0 = (const float*)d_in[11];
  const float* b1_0 = (const float*)d_in[12];
  const float* w1_1 = (const float*)d_in[13];
  const float* g1_1 = (const float*)d_in[14];
  const float* b1_1 = (const float*)d_in[15];
  const float* fc_w = (const float*)d_in[16];
  const float* fc_g = (const float*)d_in[17];
  const float* fc_b = (const float*)d_in[18];
  const float* off_w = (const float*)d_in[19];
  const float* off_b = (const float*)d_in[20];
  float* out = (float*)d_out;

  // workspace layout (all offsets 32B aligned)
  const int NQ = Bn * Mn;                                   // 4096
  int*    idx0 = (int*)d_ws;                                // NQ*16
  int*    idx1 = idx0 + NQ * S0n;                           // NQ*32
  float*  fbuf = (float*)(idx1 + NQ * S1n);                 // NQ*256 f32
  __bf16* wp00 = (__bf16*)(fbuf + (size_t)NQ * 256);        // 64*64   packed frags
  __bf16* w01  = wp00 + 64 * 64;                            // 64*128  packed frags
  __bf16* wp10 = w01 + 64 * 128;                            // 64*64   packed frags
  __bf16* w11  = wp10 + 64 * 64;                            // 64*128  packed frags
  __bf16* fcw  = w11 + 64 * 128;                            // 256*128 packed frags

  prep_weights_kernel<<<dim3(128), dim3(256), 0, stream>>>(
      w0_0, w0_1, w1_0, w1_1, fc_w, wp00, w01, wp10, w11, fcw);

  ballquery_kernel<<<dim3(NQ / 4), dim3(128), 0, stream>>>(xyz, new_xyz, idx0, idx1);

  sa_mlp_kernel<<<dim3(NQ * 2 / 4), dim3(128), 0, stream>>>(
      xyz, new_xyz, features, score, idx0, idx1,
      wp00, w01, wp10, w11,
      g0_0, b0_0, g0_1, b0_1, g1_0, b1_0, g1_1, b1_1, fbuf);

  fc_kernel<<<dim3(NQ / 16 / 4), dim3(128), 0, stream>>>(
      fbuf, fcw, fc_g, fc_b, off_w, off_b, new_xyz, out);
}
